// SelfAttention_90091234000944
// MI455X (gfx1250) — matrix operand order repaired and checked by reading, not compiled, not run on hardware
//
#include <hip/hip_runtime.h>
#include <stdint.h>

// ---------------------------------------------------------------------------
// Self-attention forward for MI455X (gfx1250, wave32, WMMA 16x16x32 bf16).
// B=4, S=2048, D=1024, NH=16, HD=64.
// Pipeline:
//   qkv_gemm x3 : fp32 GEMM via bf16 WMMA -> Q (pre-scaled), K, V^T in ws (bf16)
//   attn        : flash-attention streaming, double-buffered async-to-LDS
//                 staging, online softmax, fp32 output
// ---------------------------------------------------------------------------

typedef __attribute__((ext_vector_type(16))) __bf16        v16bf;
typedef __attribute__((ext_vector_type(8)))  float         v8f;
typedef __attribute__((ext_vector_type(8)))  unsigned int  v8u;
typedef __attribute__((ext_vector_type(4)))  int           v4i;

#define NHEADS 16
#define SEQ    2048
#define DMODEL 1024
#define HDIM   64

// ---- CDNA5 async global->LDS path (guarded; fallback = sync copy) ---------
#if defined(__has_builtin)
#  if __has_builtin(__builtin_amdgcn_global_load_async_to_lds_b128) && \
      __has_builtin(__builtin_amdgcn_s_wait_asynccnt)
#    define HAVE_ASYNC_LDS 1
#  endif
#endif
#ifndef HAVE_ASYNC_LDS
#  define HAVE_ASYNC_LDS 0
#endif

// Builtin signature (from hipcc diagnostics): param 0 is int4 in AS1 (global),
// param 1 is int4 in AS3 (LDS), then imm offset + imm cpol.
typedef __attribute__((address_space(1))) v4i as1_v4i;
typedef __attribute__((address_space(3))) v4i as3_v4i;

__device__ __forceinline__ as1_v4i* to_gbl(const void* p) {
  // global generic address == global address; integer round-trip avoids
  // addrspacecast legality questions
  return (as1_v4i*)(unsigned long long)p;
}
__device__ __forceinline__ as3_v4i* to_lds(const void* p) {
  // LDS generic address: low 32 bits == LDS byte offset (ISA §10.2 aperture)
  return (as3_v4i*)(unsigned int)(unsigned long long)p;
}

__device__ __forceinline__ unsigned short f2bf(float f) {
  unsigned int u = __builtin_bit_cast(unsigned int, f);
  u += 0x7FFFu + ((u >> 16) & 1u);           // round-to-nearest-even
  return (unsigned short)(u >> 16);
}

__device__ __forceinline__ uint4 ld4(const unsigned short* p) {
  return *(const uint4*)p;                    // 16B (caller guarantees alignment)
}

__device__ __forceinline__ v16bf frag16(uint4 a, uint4 b) {
  v8u u;
  u[0] = a.x; u[1] = a.y; u[2] = a.z; u[3] = a.w;
  u[4] = b.x; u[5] = b.y; u[6] = b.z; u[7] = b.w;
  return __builtin_bit_cast(v16bf, u);
}

__device__ __forceinline__ uint4 pack8(float4 a, float4 b) {
  uint4 r;
  r.x = (unsigned)f2bf(a.x) | ((unsigned)f2bf(a.y) << 16);
  r.y = (unsigned)f2bf(a.z) | ((unsigned)f2bf(a.w) << 16);
  r.z = (unsigned)f2bf(b.x) | ((unsigned)f2bf(b.y) << 16);
  r.w = (unsigned)f2bf(b.z) | ((unsigned)f2bf(b.w) << 16);
  return r;
}

// ---------------------------------------------------------------------------
// Kernel 1: out[m,n] = sum_k X[m,k] * W[n,k] + bias[n]   (M=8192, N=K=1024)
// mode 0/1: store bf16 as [B,NH,S,HD] (Q gets scale=1/8 folded in)
// mode 2  : store bf16 as [B,NH,HD,S] (V transposed -> contiguous for PV GEMM)
// WG: 256 thr = 8 waves (4 M x 2 N), wave tile 32x64, block tile 128x128.
// ---------------------------------------------------------------------------
__global__ __launch_bounds__(256) void qkv_gemm(
    const float* __restrict__ X, const float* __restrict__ W,
    const float* __restrict__ bias, unsigned short* __restrict__ out,
    int mode, float scale)
{
  __shared__ unsigned short As[128 * 40];  // 128 rows x 32 k, pad LD=40 (bank-safe)
  __shared__ unsigned short Bs[128 * 40];

  const int tid  = threadIdx.x;
  const int lane = tid & 31;
  const int wave = tid >> 5;
  const int half = lane >> 4;
  const int ln   = lane & 15;
  const int wm   = wave >> 1;               // 0..3
  const int wn   = wave & 1;                // 0..1
  const int m0   = blockIdx.y * 128;
  const int n0   = blockIdx.x * 128;

  const int srow = tid >> 1;                // 0..127 (staging row)
  const int skh  = (tid & 1) * 16;          // k sub-chunk: 0 or 16 elems

  v8f acc[2][4];
  #pragma unroll
  for (int i = 0; i < 2; ++i)
    #pragma unroll
    for (int j = 0; j < 4; ++j) acc[i][j] = v8f{};

  for (int k0 = 0; k0 < DMODEL; k0 += 32) {
    // stage A (X) and B (W) fp32 -> bf16 -> LDS
    {
      const float* ga = X + (size_t)(m0 + srow) * DMODEL + k0 + skh;
      float4 a0 = ((const float4*)ga)[0], a1 = ((const float4*)ga)[1];
      float4 a2 = ((const float4*)ga)[2], a3 = ((const float4*)ga)[3];
      *(uint4*)&As[srow * 40 + skh    ] = pack8(a0, a1);
      *(uint4*)&As[srow * 40 + skh + 8] = pack8(a2, a3);

      const float* gb = W + (size_t)(n0 + srow) * DMODEL + k0 + skh;
      float4 b0 = ((const float4*)gb)[0], b1 = ((const float4*)gb)[1];
      float4 b2 = ((const float4*)gb)[2], b3 = ((const float4*)gb)[3];
      *(uint4*)&Bs[srow * 40 + skh    ] = pack8(b0, b1);
      *(uint4*)&Bs[srow * 40 + skh + 8] = pack8(b2, b3);

      if (k0 + 32 < DMODEL) {                // pull next k-chunk into cache
        __builtin_prefetch(ga + 32, 0, 0);   // -> global_prefetch_b8
        __builtin_prefetch(gb + 32, 0, 0);
      }
    }
    __syncthreads();

    v16bf af[2], bfr[4];
    #pragma unroll
    for (int mi = 0; mi < 2; ++mi) {
      const unsigned short* p = &As[(wm * 32 + mi * 16 + ln) * 40];
      af[mi] = frag16(ld4(p + 8 * half), ld4(p + 16 + 8 * half));
    }
    #pragma unroll
    for (int ni = 0; ni < 4; ++ni) {
      const unsigned short* p = &Bs[(wn * 64 + ni * 16 + ln) * 40];
      bfr[ni] = frag16(ld4(p + 8 * half), ld4(p + 16 + 8 * half));
    }
    #pragma unroll
    for (int mi = 0; mi < 2; ++mi)
      #pragma unroll
      for (int ni = 0; ni < 4; ++ni)
        acc[mi][ni] = __builtin_amdgcn_wmma_f32_16x16x32_bf16(
            false, af[mi], false, bfr[ni], (short)0, acc[mi][ni], false, false);
    __syncthreads();
  }

  float bn[4];
  #pragma unroll
  for (int ni = 0; ni < 4; ++ni) bn[ni] = bias[n0 + wn * 64 + ni * 16 + ln];

  if (mode < 2) {
    // [B, NH, S, HD] layout: per-element b16 stores
    #pragma unroll
    for (int mi = 0; mi < 2; ++mi) {
      #pragma unroll
      for (int ni = 0; ni < 4; ++ni) {
        const int n  = n0 + wn * 64 + ni * 16 + ln;
        const int hh = n >> 6, dd = n & 63;
        #pragma unroll
        for (int r = 0; r < 8; ++r) {
          const int m  = m0 + wm * 32 + mi * 16 + 8 * half + r;
          const int bb = m >> 11, ss = m & (SEQ - 1);
          const float v = (acc[mi][ni][r] + bn[ni]) * scale;
          out[(((size_t)bb * NHEADS + hh) * SEQ + ss) * HDIM + dd] = f2bf(v);
        }
      }
    }
  } else {
    // V^T: [B, NH, HD, S] layout -> 8 consecutive s per lane = one b128 store
    #pragma unroll
    for (int mi = 0; mi < 2; ++mi) {
      const int mb = m0 + wm * 32 + mi * 16 + 8 * half;
      const int bb = mb >> 11, ss = mb & (SEQ - 1);
      #pragma unroll
      for (int ni = 0; ni < 4; ++ni) {
        const int n  = n0 + wn * 64 + ni * 16 + ln;
        const int hh = n >> 6, dd = n & 63;
        float v[8];
        #pragma unroll
        for (int r = 0; r < 8; ++r) v[r] = (acc[mi][ni][r] + bn[ni]) * scale;
        uint4 pk;
        pk.x = (unsigned)f2bf(v[0]) | ((unsigned)f2bf(v[1]) << 16);
        pk.y = (unsigned)f2bf(v[2]) | ((unsigned)f2bf(v[3]) << 16);
        pk.z = (unsigned)f2bf(v[4]) | ((unsigned)f2bf(v[5]) << 16);
        pk.w = (unsigned)f2bf(v[6]) | ((unsigned)f2bf(v[7]) << 16);
        *(uint4*)&out[(((size_t)bb * NHEADS + hh) * HDIM + dd) * SEQ + ss] = pk;
      }
    }
  }
}

// ---------------------------------------------------------------------------
// Kernel 2: streaming attention. WG = 128 thr (4 waves); each wave owns 16
// query rows; iterate over 64-wide key blocks with online softmax.
// K/V^T blocks double-buffered in LDS via GLOBAL_LOAD_ASYNC_TO_LDS_B128
// (ASYNCcnt-tracked) so next block's copy overlaps current block's WMMAs.
// Q already scaled by 1/sqrt(HD).  Output fp32 [B, S, D].
// ---------------------------------------------------------------------------
__global__ __launch_bounds__(128) void attn(
    const unsigned short* __restrict__ Qb,   // [BH, S, HD] bf16
    const unsigned short* __restrict__ Kb,   // [BH, S, HD] bf16
    const unsigned short* __restrict__ Vt,   // [BH, HD, S] bf16
    float* __restrict__ out)                 // [B, S, D] fp32
{
  __shared__ unsigned short Ks[2][64 * 72];  // rows = key s (64), cols = d, LD=72
  __shared__ unsigned short Vs[2][64 * 72];  // rows = d (64), cols = key s
  __shared__ unsigned short Ps[4][16 * 72];  // per-wave P tile (16 x 64)

  const int tid  = threadIdx.x;
  const int lane = tid & 31;
  const int w    = tid >> 5;
  const int half = lane >> 4;
  const int ln   = lane & 15;
  const int bh   = blockIdx.y;               // b*NH + h
  const int b    = bh >> 4, h = bh & 15;
  const int q0   = blockIdx.x * 64;

  const int srow = tid >> 1;                 // 0..63
  const int sch  = (tid & 1) * 32;           // col half (elems)

  // issue staging of key-block starting at element kbase into buffer buf
  auto stage = [&](int kbase, int buf) {
    const unsigned short* kg = Kb + ((size_t)bh * SEQ + kbase + srow) * HDIM + sch;
    const unsigned short* vg = Vt + ((size_t)bh * HDIM + srow) * SEQ + kbase + sch;
    unsigned short* ksl = &Ks[buf][srow * 72 + sch];
    unsigned short* vsl = &Vs[buf][srow * 72 + sch];
#if HAVE_ASYNC_LDS
    #pragma unroll
    for (int i = 0; i < 4; ++i) {
      __builtin_amdgcn_global_load_async_to_lds_b128(
          to_gbl(kg + i * 8), to_lds(ksl + i * 8), 0, 0);
      __builtin_amdgcn_global_load_async_to_lds_b128(
          to_gbl(vg + i * 8), to_lds(vsl + i * 8), 0, 0);
    }
#else
    #pragma unroll
    for (int i = 0; i < 4; ++i) {
      *(uint4*)(ksl + i * 8) = *(const uint4*)(kg + i * 8);
      *(uint4*)(vsl + i * 8) = *(const uint4*)(vg + i * 8);
    }
#endif
  };

  // Q fragments (16 rows x HD=64, two k-steps of 32) kept in registers
  v16bf qf[2];
  {
    const unsigned short* qp = Qb + ((size_t)bh * SEQ + q0 + w * 16 + ln) * HDIM;
    #pragma unroll
    for (int ks = 0; ks < 2; ++ks)
      qf[ks] = frag16(ld4(qp + ks * 32 + 8 * half),
                      ld4(qp + ks * 32 + 16 + 8 * half));
  }

  v8f O[4];
  #pragma unroll
  for (int dt = 0; dt < 4; ++dt) O[dt] = v8f{};
  float mr[8], lr[8];
  #pragma unroll
  for (int r = 0; r < 8; ++r) { mr[r] = -3.0e38f; lr[r] = 0.0f; }

  // prologue: fill buffer 0
  stage(0, 0);
#if HAVE_ASYNC_LDS
  __builtin_amdgcn_s_wait_asynccnt(0);
#endif
  __syncthreads();

  const int NKB = SEQ / 64;                  // 32 key blocks
  for (int kb = 0; kb < NKB; ++kb) {
    const int cur = kb & 1;

    // overlap: kick off next block's async copy before computing
    if (kb + 1 < NKB) stage((kb + 1) * 64, cur ^ 1);

    // S = Q * K^T  (4 column-tiles of 16, k-dim = HD = 64 -> 2 WMMAs each)
    v8f sacc[4];
    #pragma unroll
    for (int nt = 0; nt < 4; ++nt) {
      v8f c = v8f{};
      #pragma unroll
      for (int ks = 0; ks < 2; ++ks) {
        const unsigned short* p = &Ks[cur][(nt * 16 + ln) * 72 + ks * 32 + 8 * half];
        v16bf bb = frag16(ld4(p), ld4(p + 16));
        c = __builtin_amdgcn_wmma_f32_16x16x32_bf16(
            false, qf[ks], false, bb, (short)0, c, false, false);
      }
      sacc[nt] = c;
    }

    // online softmax per row (rows r+8*half live in the 16 lanes of a half)
    #pragma unroll
    for (int r = 0; r < 8; ++r) {
      float mx = fmaxf(fmaxf(sacc[0][r], sacc[1][r]),
                       fmaxf(sacc[2][r], sacc[3][r]));
      mx = fmaxf(mx, __shfl_xor(mx, 1, 32));
      mx = fmaxf(mx, __shfl_xor(mx, 2, 32));
      mx = fmaxf(mx, __shfl_xor(mx, 4, 32));
      mx = fmaxf(mx, __shfl_xor(mx, 8, 32));
      const float mn = fmaxf(mr[r], mx);
      const float al = __expf(mr[r] - mn);
      float s = 0.0f;
      #pragma unroll
      for (int nt = 0; nt < 4; ++nt) {
        const float p = __expf(sacc[nt][r] - mn);
        sacc[nt][r] = p;
        s += p;
      }
      s += __shfl_xor(s, 1, 32);
      s += __shfl_xor(s, 2, 32);
      s += __shfl_xor(s, 4, 32);
      s += __shfl_xor(s, 8, 32);
      lr[r] = lr[r] * al + s;
      mr[r] = mn;
      #pragma unroll
      for (int dt = 0; dt < 4; ++dt) O[dt][r] *= al;
    }

    // C-layout -> A-layout: spill P (bf16) to wave-private LDS
    #pragma unroll
    for (int nt = 0; nt < 4; ++nt)
      #pragma unroll
      for (int r = 0; r < 8; ++r)
        Ps[w][(8 * half + r) * 72 + nt * 16 + ln] = f2bf(sacc[nt][r]);

    // O += P * V^T  (k-dim = 64 keys -> 2 steps; 4 d-tiles)
    #pragma unroll
    for (int ks = 0; ks < 2; ++ks) {
      const unsigned short* pp = &Ps[w][ln * 72 + ks * 32 + 8 * half];
      v16bf pf = frag16(ld4(pp), ld4(pp + 16));
      #pragma unroll
      for (int dt = 0; dt < 4; ++dt) {
        const unsigned short* vp = &Vs[cur][(dt * 16 + ln) * 72 + ks * 32 + 8 * half];
        v16bf bb = frag16(ld4(vp), ld4(vp + 16));
        O[dt] = __builtin_amdgcn_wmma_f32_16x16x32_bf16(
            false, pf, false, bb, (short)0, O[dt], false, false);
      }
    }

    // next buffer must be fully written, and everyone done with cur
#if HAVE_ASYNC_LDS
    __builtin_amdgcn_s_wait_asynccnt(0);
#endif
    __syncthreads();
  }

  // epilogue: O / l -> fp32 out [B, S, D]
  #pragma unroll
  for (int r = 0; r < 8; ++r) {
    const float inv = 1.0f / lr[r];
    const int ss = q0 + w * 16 + 8 * half + r;
    float* op = out + ((size_t)b * SEQ + ss) * DMODEL + h * HDIM;
    #pragma unroll
    for (int dt = 0; dt < 4; ++dt)
      op[dt * 16 + ln] = O[dt][r] * inv;
  }
}

// ---------------------------------------------------------------------------
extern "C" void kernel_launch(void* const* d_in, const int* in_sizes, int n_in,
                              void* d_out, int out_size, void* d_ws, size_t ws_size,
                              hipStream_t stream) {
  (void)in_sizes; (void)n_in; (void)out_size; (void)ws_size;
  const float* hs = (const float*)d_in[0];
  const float* Wq = (const float*)d_in[1];
  const float* bq = (const float*)d_in[2];
  const float* Wk = (const float*)d_in[3];
  const float* bk = (const float*)d_in[4];
  const float* Wv = (const float*)d_in[5];
  const float* bv = (const float*)d_in[6];
  float* outp = (float*)d_out;

  const size_t elems = (size_t)4 * SEQ * DMODEL;   // 8192*1024
  unsigned short* Qb = (unsigned short*)d_ws;
  unsigned short* Kb = Qb + elems;
  unsigned short* Vt = Kb + elems;

  dim3 g1(DMODEL / 128, (4 * SEQ) / 128);          // (8, 64)
  dim3 b1(256);
  qkv_gemm<<<g1, b1, 0, stream>>>(hs, Wq, bq, Qb, 0, 0.125f);  // 1/sqrt(64) folded
  qkv_gemm<<<g1, b1, 0, stream>>>(hs, Wk, bk, Kb, 1, 1.0f);
  qkv_gemm<<<g1, b1, 0, stream>>>(hs, Wv, bv, Vt, 2, 1.0f);    // stored transposed

  dim3 g2(SEQ / 64, 4 * NHEADS);                   // (32, 64)
  dim3 b2(128);
  attn<<<g2, b2, 0, stream>>>(Qb, Kb, Vt, outp);
}
